// Decoder_49323404427578
// MI455X (gfx1250) — matrix operand order repaired and checked by reading, not compiled, not run on hardware
//
#include <hip/hip_runtime.h>
#include <hip/hip_bf16.h>

typedef __attribute__((ext_vector_type(16))) _Float16 v16h;
typedef __attribute__((ext_vector_type(8)))  _Float16 v8h;
typedef __attribute__((ext_vector_type(8)))  float    v8f;

#define BB 64
#define LL 196
#define DD 512
#define AAD 512
#define MMD 512
#define NND 512
#define KV 10000
#define TSTEPS 31
#define TMAX 32

__device__ __forceinline__ float sigf(float x) { return 1.0f / (1.0f + __expf(-x)); }

__device__ __forceinline__ v16h cat8(v8h lo, v8h hi) {
    return __builtin_shufflevector(lo, hi, 0, 1, 2, 3, 4, 5, 6, 7,
                                   8, 9, 10, 11, 12, 13, 14, 15);
}

// ---------------------------------------------------------------------------
// WMMA GEMM with 4-way M register blocking:
//   each wave computes a 64x16 strip of C (4 accumulators). Per k-chunk all
//   A fragments are loaded as b128 pairs, B as 16 strided f16 loads (one clause, staged waits), then
//   4 back-to-back v_wmma_f32_16x16x32_f16 issues share the single B fragment.
// C(MxN) = act(A(MxK,f16) * B(KxN,f16) + bias + extra) [* row mask]
// Requires M % 64 == 0, N % 16 == 0, K % 32 == 0; rows 16B aligned.
// ---------------------------------------------------------------------------
template <int MT>
__global__ __launch_bounds__(128)
void gemm_wmma(const _Float16* __restrict__ A, int lda,
               const _Float16* __restrict__ Bm, int ldb,
               const float* __restrict__ bias,
               const float* __restrict__ extra, int lde,
               float* __restrict__ Cf, int ldc,
               _Float16* __restrict__ Ch, int ldch,
               int M, int N, int K, int act,
               const int* __restrict__ cap_len, int t)
{
    const int lane   = threadIdx.x;                 // 0..31 (wave32)
    const int tilesN = N >> 4;
    const int wid    = blockIdx.x * 4 + threadIdx.y;
    if (wid >= (M / (16 * MT)) * tilesN) return;
    const int tmg = wid / tilesN, tn = wid % tilesN;
    const int m0 = tmg * 16 * MT, n0 = tn << 4;

    v8f acc[MT];
    #pragma unroll
    for (int i = 0; i < MT; ++i) acc[i] = (v8f){0.f,0.f,0.f,0.f,0.f,0.f,0.f,0.f};

    // A fragment rows: lane 0-15 -> M=base+lane, K = k0 + [0..7,16..23]
    //                  lane 16-31 -> M=base+lane-16, K = k0+8+[0..7,16..23]
    const _Float16* arow[MT];
    #pragma unroll
    for (int i = 0; i < MT; ++i)
        arow[i] = A + (size_t)(m0 + i * 16 + (lane & 15)) * lda + ((lane >> 4) << 3);
    // B fragment: lane 0-15 -> N=n0+lane, K = k0 + [0..7,16..23]; lane 16-31 -> K = k0+8+[0..7,16..23]
    const _Float16* bcol = Bm + (size_t)((lane >> 4) << 3) * ldb + n0 + (lane & 15);

    for (int k0 = 0; k0 < K; k0 += 32) {
        // ---- A fragment loads as b128 pairs, B as strided f16 loads (clause + staged waits) ----
        const _Float16* bp = bcol + (size_t)k0 * ldb;
        v16h b;
        #pragma unroll
        for (int s = 0; s < 16; ++s) b[s] = bp[(size_t)(s + ((s >> 3) << 3)) * ldb];

        v16h a[MT];
        #pragma unroll
        for (int i = 0; i < MT; ++i)
            a[i] = cat8(*(const v8h*)(arow[i] + k0),
                        *(const v8h*)(arow[i] + k0 + 16));

        if (k0 + 32 < K) __builtin_prefetch(bp + (size_t)32 * ldb, 0, 3);

        // ---- 4 back-to-back WMMAs sharing the B fragment ----
        #pragma unroll
        for (int i = 0; i < MT; ++i)
            acc[i] = __builtin_amdgcn_wmma_f32_16x16x32_f16(false, a[i], false, b,
                                                            (short)0, acc[i],
                                                            false, false);
    }

    // C layout per tile: lane 0-15 -> N=n0+lane, rows base..base+7 in acc[0..7]
    //                    lane 16-31 -> N=n0+lane-16, rows base+8..base+15
    const int cn = n0 + (lane & 15);
    const float bz = bias ? bias[cn] : 0.f;
    #pragma unroll
    for (int i = 0; i < MT; ++i) {
        const int mb = m0 + i * 16 + ((lane >> 4) << 3);
        #pragma unroll
        for (int r = 0; r < 8; ++r) {
            const int cm = mb + r;
            float v = acc[i][r] + bz;
            if (extra) v += extra[(size_t)cm * lde + cn];
            if (act == 1) v = tanhf(v);
            else if (act == 2) v = sigf(v);
            if (cap_len) v *= (t < cap_len[cm] - 1) ? 1.0f : 0.0f;
            if (Cf) Cf[(size_t)cm * ldc + cn] = v;
            if (Ch) Ch[(size_t)cm * ldch + cn] = (_Float16)v;
        }
    }
}

// ---------------------------------------------------------------------------
// Elementwise helpers
// ---------------------------------------------------------------------------
__global__ void f2h_kernel(const float* __restrict__ a, _Float16* __restrict__ o, int n) {
    for (int i = blockIdx.x * blockDim.x + threadIdx.x; i < n; i += gridDim.x * blockDim.x)
        o[i] = (_Float16)a[i];
}

__global__ void addvec_kernel(const float* __restrict__ a, const float* __restrict__ b,
                              float* __restrict__ o, int n) {
    int i = blockIdx.x * blockDim.x + threadIdx.x;
    if (i < n) o[i] = a[i] + b[i];
}

__global__ void mean_enc_kernel(const float* __restrict__ enc, _Float16* __restrict__ out) {
    int idx = blockIdx.x * blockDim.x + threadIdx.x;
    if (idx >= BB * DD) return;
    int b = idx / DD, d = idx % DD;
    const float* p = enc + (size_t)b * LL * DD + d;
    float s = 0.f;
    for (int l = 0; l < LL; ++l) s += p[(size_t)l * DD];
    out[idx] = (_Float16)(s * (1.0f / LL));
}

__global__ void emb_kernel(const float* __restrict__ emb, const int* __restrict__ captions,
                           int t, _Float16* __restrict__ xbuf, float* __restrict__ embf) {
    int idx = blockIdx.x * blockDim.x + threadIdx.x;
    if (idx >= BB * MMD) return;
    int b = idx / MMD, d = idx % MMD;
    int tok = captions[b * TMAX + t];
    float v = emb[(size_t)tok * MMD + d];
    embf[idx] = v;
    xbuf[b * 1536 + d] = (_Float16)v;
}

// ---------------------------------------------------------------------------
// Attention: e = relu(att1 + att2) . w_fa ; softmax over L ; z = beta * (alpha @ enc)
// One block per batch row.
// ---------------------------------------------------------------------------
__global__ __launch_bounds__(256)
void att_kernel(const float* __restrict__ att1, const float* __restrict__ att2,
                const float* __restrict__ w_fa, const float* __restrict__ b_fa,
                const float* __restrict__ enc, const float* __restrict__ betab,
                const int* __restrict__ cap_len, int t,
                float* __restrict__ alphas_out,
                _Float16* __restrict__ xbuf, _Float16* __restrict__ hzbuf)
{
    __shared__ float s_att2[AAD];
    __shared__ float s_wfa[AAD];
    __shared__ float s_e[LL];
    __shared__ float s_red[256];
    const int b = blockIdx.x;
    const int tid = threadIdx.x;

    for (int i = tid; i < AAD; i += 256) {
        s_att2[i] = att2[b * AAD + i];
        s_wfa[i]  = w_fa[i];
    }
    __syncthreads();

    const float bfa = b_fa[0];
    for (int l = tid; l < LL; l += 256) {
        const float* a1 = att1 + ((size_t)b * LL + l) * AAD;
        float acc = bfa;
        for (int a = 0; a < AAD; ++a) {
            float r = a1[a] + s_att2[a];
            acc += (r > 0.f ? r : 0.f) * s_wfa[a];
        }
        s_e[l] = acc;
    }
    __syncthreads();

    // block softmax over L=196
    float v = (tid < LL) ? s_e[tid] : -3.4e38f;
    s_red[tid] = v; __syncthreads();
    for (int s = 128; s > 0; s >>= 1) {
        if (tid < s) s_red[tid] = fmaxf(s_red[tid], s_red[tid + s]);
        __syncthreads();
    }
    const float mx = s_red[0];
    __syncthreads();
    float ex = (tid < LL) ? __expf(s_e[tid] - mx) : 0.f;
    s_red[tid] = ex; __syncthreads();
    for (int s = 128; s > 0; s >>= 1) {
        if (tid < s) s_red[tid] += s_red[tid + s];
        __syncthreads();
    }
    const float inv = 1.0f / s_red[0];
    __syncthreads();
    if (tid < LL) s_e[tid] = ex * inv;          // alpha
    __syncthreads();

    const float mf = (t < cap_len[b] - 1) ? 1.0f : 0.0f;
    if (tid < LL)
        alphas_out[((size_t)b * TSTEPS + t) * LL + tid] = s_e[tid] * mf;

    for (int d = tid; d < DD; d += 256) {
        const float* ep = enc + (size_t)b * LL * DD + d;
        float acc = 0.f;
        for (int l = 0; l < LL; ++l) acc += s_e[l] * ep[(size_t)l * DD];
        float z = acc * betab[b * DD + d];
        xbuf[b * 1536 + 512 + d]  = (_Float16)z;   // LSTM input slot
        hzbuf[b * 1024 + 512 + d] = (_Float16)z;   // deep-output slot
    }
}

// ---------------------------------------------------------------------------
// LSTM cell + ragged-mask state update (h kept as f16 state in xbuf, c in f32)
// ---------------------------------------------------------------------------
__global__ void lstm_kernel(const float* __restrict__ gates, float* __restrict__ c,
                            _Float16* __restrict__ xbuf, _Float16* __restrict__ hzbuf,
                            const int* __restrict__ cap_len, int t)
{
    int idx = blockIdx.x * blockDim.x + threadIdx.x;
    if (idx >= BB * NND) return;
    int b = idx / NND, n = idx % NND;
    const float* g = gates + (size_t)b * 4 * NND;
    float ig = sigf(g[n]);
    float fg = sigf(g[NND + n]);
    float gg = tanhf(g[2 * NND + n]);
    float og = sigf(g[3 * NND + n]);
    float cn = fg * c[idx] + ig * gg;
    float hn = og * tanhf(cn);
    hzbuf[b * 1024 + n] = (_Float16)hn;          // h_new always feeds deep output
    if (t < cap_len[b] - 1) {                    // masked recurrent update
        c[idx] = cn;
        xbuf[b * 1536 + 1024 + n] = (_Float16)hn;
    }
}

// ---------------------------------------------------------------------------
// Host
// ---------------------------------------------------------------------------
static inline void launch_gemm(hipStream_t s, const _Float16* A, int lda,
                               const _Float16* B, int ldb, const float* bias,
                               const float* extra, int lde,
                               float* Cf, int ldc, _Float16* Ch, int ldch,
                               int M, int N, int K, int act,
                               const int* cl, int t)
{
    int waves = (M / 64) * (N / 16);       // one wave per 64x16 strip
    dim3 blk(32, 4);
    int grid = (waves + 3) / 4;
    gemm_wmma<4><<<grid, blk, 0, s>>>(A, lda, B, ldb, bias, extra, lde,
                                      Cf, ldc, Ch, ldch, M, N, K, act, cl, t);
}

extern "C" void kernel_launch(void* const* d_in, const int* in_sizes, int n_in,
                              void* d_out, int out_size, void* d_ws, size_t ws_size,
                              hipStream_t stream)
{
    (void)in_sizes; (void)n_in; (void)out_size; (void)ws_size;
    const float* enc     = (const float*)d_in[0];
    const int*   caps    = (const int*)  d_in[1];
    const int*   cap_len = (const int*)  d_in[2];
    const float* emb     = (const float*)d_in[3];
    const float* W_ea = (const float*)d_in[4];  const float* b_ea = (const float*)d_in[5];
    const float* W_da = (const float*)d_in[6];  const float* b_da = (const float*)d_in[7];
    const float* w_fa = (const float*)d_in[8];  const float* b_fa = (const float*)d_in[9];
    const float* W_hi = (const float*)d_in[10]; const float* b_hi = (const float*)d_in[11];
    const float* W_ci = (const float*)d_in[12]; const float* b_ci = (const float*)d_in[13];
    const float* W_ih = (const float*)d_in[14]; const float* W_hh = (const float*)d_in[15];
    const float* b_ih = (const float*)d_in[16]; const float* b_hh = (const float*)d_in[17];
    const float* W_fb = (const float*)d_in[18]; const float* b_fb = (const float*)d_in[19];
    const float* W_lh = (const float*)d_in[20]; const float* b_lh = (const float*)d_in[21];
    const float* W_lz = (const float*)d_in[22]; const float* b_lz = (const float*)d_in[23];
    const float* W_lo = (const float*)d_in[24]; const float* b_lo = (const float*)d_in[25];

    float* pred = (float*)d_out;                       // (B, T, K)
    float* alph = pred + (size_t)BB * TSTEPS * KV;     // (B, T, L)

    // ---- carve workspace ----
    char* p = (char*)d_ws;
    auto alloc = [&](size_t bytes) -> void* {
        void* r = (void*)p;
        p += (bytes + 255) & ~(size_t)255;
        return r;
    };
    _Float16* enc_h  = (_Float16*)alloc((size_t)BB * LL * DD * 2);
    float*    att1   = (float*)   alloc((size_t)BB * LL * AAD * 4);
    _Float16* Wea_h  = (_Float16*)alloc((size_t)DD * AAD * 2);
    _Float16* Wda_h  = (_Float16*)alloc((size_t)NND * AAD * 2);
    _Float16* Wfb_h  = (_Float16*)alloc((size_t)NND * DD * 2);
    _Float16* Whi_h  = (_Float16*)alloc((size_t)DD * NND * 2);
    _Float16* Wci_h  = (_Float16*)alloc((size_t)DD * NND * 2);
    _Float16* Wcat_h = (_Float16*)alloc((size_t)1536 * 2048 * 2);   // [W_ih; W_hh]
    _Float16* WL_h   = (_Float16*)alloc((size_t)1024 * MMD * 2);    // [W_lh; W_lz]
    _Float16* Wlo_h  = (_Float16*)alloc((size_t)MMD * KV * 2);
    _Float16* mean_h = (_Float16*)alloc((size_t)BB * DD * 2);
    float*    bg     = (float*)   alloc(2048 * 4);                  // b_ih + b_hh
    float*    bs     = (float*)   alloc(512 * 4);                   // b_lh + b_lz
    float*    att2   = (float*)   alloc((size_t)BB * AAD * 4);
    float*    betab  = (float*)   alloc((size_t)BB * DD * 4);
    float*    gates  = (float*)   alloc((size_t)BB * 2048 * 4);
    _Float16* xbuf   = (_Float16*)alloc((size_t)BB * 1536 * 2);     // [emb | z | h]
    _Float16* hzbuf  = (_Float16*)alloc((size_t)BB * 1024 * 2);     // [h_new | z]
    float*    cbuf   = (float*)   alloc((size_t)BB * NND * 4);
    float*    embf   = (float*)   alloc((size_t)BB * MMD * 4);
    _Float16* s_h    = (_Float16*)alloc((size_t)BB * MMD * 2);

    auto f2h = [&](const float* a, _Float16* o, int n) {
        f2h_kernel<<<(n + 255) / 256, 256, 0, stream>>>(a, o, n);
    };

    // ---- phase 0: fp16 weight prep ----
    f2h(enc,  enc_h, BB * LL * DD);
    f2h(W_ea, Wea_h, DD * AAD);
    f2h(W_da, Wda_h, NND * AAD);
    f2h(W_fb, Wfb_h, NND * DD);
    f2h(W_hi, Whi_h, DD * NND);
    f2h(W_ci, Wci_h, DD * NND);
    f2h(W_ih, Wcat_h,                 1024 * 2048);
    f2h(W_hh, Wcat_h + 1024 * 2048,    512 * 2048);
    f2h(W_lh, WL_h,                    512 * MMD);
    f2h(W_lz, WL_h + 512 * MMD,        512 * MMD);
    f2h(W_lo, Wlo_h, MMD * KV);
    addvec_kernel<<<(2048 + 255) / 256, 256, 0, stream>>>(b_ih, b_hh, bg, 2048);
    addvec_kernel<<<(512  + 255) / 256, 256, 0, stream>>>(b_lh, b_lz, bs, 512);
    mean_enc_kernel<<<(BB * DD + 255) / 256, 256, 0, stream>>>(enc, mean_h);

    // att1 = enc @ W_ea + b_ea   (12544 x 512 x 512)
    launch_gemm(stream, enc_h, DD, Wea_h, AAD, b_ea, nullptr, 0,
                att1, AAD, nullptr, 0, BB * LL, AAD, DD, 0, nullptr, 0);
    // h0 = tanh(mean @ W_hi + b_hi) -> f16 into xbuf h-slot
    launch_gemm(stream, mean_h, DD, Whi_h, NND, b_hi, nullptr, 0,
                nullptr, 0, xbuf + 1024, 1536, BB, NND, DD, 1, nullptr, 0);
    // c0 = tanh(mean @ W_ci + b_ci) -> f32
    launch_gemm(stream, mean_h, DD, Wci_h, NND, b_ci, nullptr, 0,
                cbuf, NND, nullptr, 0, BB, NND, DD, 1, nullptr, 0);

    // ---- sequential decode ----
    for (int t = 0; t < TSTEPS; ++t) {
        // att2 = h @ W_da + b_da
        launch_gemm(stream, xbuf + 1024, 1536, Wda_h, AAD, b_da, nullptr, 0,
                    att2, AAD, nullptr, 0, BB, AAD, NND, 0, nullptr, 0);
        // beta = sigmoid(h @ W_fb + b_fb)
        launch_gemm(stream, xbuf + 1024, 1536, Wfb_h, DD, b_fb, nullptr, 0,
                    betab, DD, nullptr, 0, BB, DD, NND, 2, nullptr, 0);
        // softmax attention + gated context z
        att_kernel<<<BB, 256, 0, stream>>>(att1, att2, w_fa, b_fa, enc, betab,
                                           cap_len, t, alph, xbuf, hzbuf);
        // embedding gather for this timestep
        emb_kernel<<<(BB * MMD + 255) / 256, 256, 0, stream>>>(emb, caps, t, xbuf, embf);
        // gates = [emb|z|h] @ [W_ih;W_hh] + (b_ih + b_hh)
        launch_gemm(stream, xbuf, 1536, Wcat_h, 2048, bg, nullptr, 0,
                    gates, 2048, nullptr, 0, BB, 2048, 1536, 0, nullptr, 0);
        // LSTM cell + masked state update
        lstm_kernel<<<(BB * NND + 255) / 256, 256, 0, stream>>>(gates, cbuf, xbuf,
                                                                hzbuf, cap_len, t);
        // s = emb + [h_new|z] @ [W_lh;W_lz] + (b_lh + b_lz)
        launch_gemm(stream, hzbuf, 1024, WL_h, MMD, bs, embf, MMD,
                    nullptr, 0, s_h, MMD, BB, MMD, 1024, 0, nullptr, 0);
        // logits = (s @ W_lo + b_lo) * mask -> directly into predictions (ldc = T*K)
        launch_gemm(stream, s_h, MMD, Wlo_h, KV, b_lo, nullptr, 0,
                    pred + (size_t)t * KV, TSTEPS * KV, nullptr, 0,
                    BB, KV, MMD, 0, cap_len, t);
    }
}
